// MultiheadDiffAttn_23991687315541
// MI455X (gfx1250) — matrix operand order repaired and checked by reading, not compiled, not run on hardware
//
#include <hip/hip_runtime.h>
#include <hip/hip_bf16.h>
#include <math.h>

// ---------------------------------------------------------------------------
// MultiheadDiffAttn for MI455X (gfx1250): bf16 WMMA everywhere, flash-fused
// differential attention, async-to-LDS staging for the K tiles.
// ---------------------------------------------------------------------------

typedef __attribute__((ext_vector_type(16))) __bf16 v16bf;
typedef __attribute__((ext_vector_type(8)))  __bf16 v8bf;
typedef __attribute__((ext_vector_type(8)))  float  v8f;

#define LAMBDA_INIT_F 0.78360576652f   // 0.8 - 0.6*exp(-0.3*12)
#define SCALING_F     0.125f           // 64^-0.5

__device__ __forceinline__ v8bf ld8(const __bf16* p) {
    return *(const v8bf*)p;
}
// A-fragment: elements 0..15 = K[half*16 .. +15]
__device__ __forceinline__ v16bf cat16(v8bf a, v8bf b) {
    return __builtin_shufflevector(a, b, 0,1,2,3,4,5,6,7,8,9,10,11,12,13,14,15);
}

// Async copy 16B global -> LDS (gfx1250 GLOBAL_LOAD_ASYNC_TO_LDS_B128).
// Same INST_OFFSET applies to both the LDS and global addresses, so one
// (lds,addr) operand pair serves two chunks via offset:0 / offset:16.
__device__ __forceinline__ void async_b128(unsigned int lds_off, const void* gaddr) {
    asm volatile("global_load_async_to_lds_b128 %0, %1, off"
                 :: "v"(lds_off), "v"(gaddr) : "memory");
}
__device__ __forceinline__ void async_b128_o16(unsigned int lds_off, const void* gaddr) {
    asm volatile("global_load_async_to_lds_b128 %0, %1, off offset:16"
                 :: "v"(lds_off), "v"(gaddr) : "memory");
}
__device__ __forceinline__ void wait_async0() {
    asm volatile("s_wait_asynccnt 0x0" ::: "memory");
}

// ---------------------------------------------------------------------------
// Generic tiled GEMM: C[MxN] = A[MxK] * B[KxN], bf16 WMMA, fp32 accumulate.
// Block = 256 threads = 8 waves, tile 128x128, K-step 64 (two wmma k-steps).
// All problem dims here are multiples of 128 -> no bounds checks.
// ---------------------------------------------------------------------------
template <typename TA, typename TC>
__global__ __launch_bounds__(256) void gemm_bf16_wmma(
    const TA* __restrict__ A, const float* __restrict__ B, TC* __restrict__ C,
    int M, int N, int K)
{
    __shared__ __align__(32) __bf16 As[128 * 64];   // [m][k] row-major
    __shared__ __align__(32) __bf16 Bt[128 * 64];   // [n][k] (transposed stage)

    const int tid  = threadIdx.x;
    const int w    = tid >> 5, lane = tid & 31;
    const int lrow = lane & 15, half = lane >> 4;
    const int wm   = w >> 2,   wn   = w & 3;        // 2x4 wave grid
    const int m0   = blockIdx.y * 128, n0 = blockIdx.x * 128;

    v8f zero = {};
    v8f acc[4][2];
#pragma unroll
    for (int i = 0; i < 4; ++i)
#pragma unroll
        for (int j = 0; j < 2; ++j) acc[i][j] = zero;

    const int arow = tid >> 1, ac0 = (tid & 1) * 32;   // A staging coords
    const int bkk  = tid >> 2, bn0 = (tid & 3) * 32;   // B staging coords

    for (int k0 = 0; k0 < K; k0 += 64) {
        // Stage A tile 128x64 (convert to bf16).
        {
            const TA* src = A + (size_t)(m0 + arow) * K + k0 + ac0;
#pragma unroll
            for (int j = 0; j < 32; ++j)
                As[arow * 64 + ac0 + j] = (__bf16)(float)src[j];
        }
        // Stage B tile 64x128 transposed -> Bt[n][k].
        {
            const float* src = B + (size_t)(k0 + bkk) * N + n0 + bn0;
#pragma unroll
            for (int j = 0; j < 32; ++j)
                Bt[(bn0 + j) * 64 + bkk] = (__bf16)src[j];
        }
        // Warm L2 for the next K tile (global_prefetch_b8).
        if (k0 + 64 < K) {
            __builtin_prefetch(A + (size_t)(m0 + arow) * K + k0 + 64 + ac0, 0, 1);
            __builtin_prefetch(B + (size_t)(k0 + 64 + bkk) * N + n0 + bn0, 0, 1);
        }
        __syncthreads();

#pragma unroll
        for (int kf = 0; kf < 2; ++kf) {
            // Batch all fragment loads, then issue the 8 independent WMMAs.
            v16bf af[4];
#pragma unroll
            for (int i = 0; i < 4; ++i) {
                const __bf16* p = &As[(wm*64 + i*16 + lrow) * 64 + kf*32 + half*16];
                af[i] = cat16(ld8(p), ld8(p + 8));
            }
            v16bf bfr[2];
#pragma unroll
            for (int j = 0; j < 2; ++j)
                bfr[j] = *(const v16bf*)&Bt[(wn*32 + j*16 + lrow) * 64 + kf*32 + half*16];
#pragma unroll
            for (int j = 0; j < 2; ++j)
#pragma unroll
                for (int i = 0; i < 4; ++i)
                    acc[i][j] = __builtin_amdgcn_wmma_f32_16x16x32_bf16(
                        false, af[i], false, bfr[j], (short)0, acc[i][j], false, false);
        }
        __syncthreads();
    }

    // Epilogue: C layout (VGPR r: m = r + 8*half, n = lane&15).
#pragma unroll
    for (int i = 0; i < 4; ++i)
#pragma unroll
        for (int j = 0; j < 2; ++j)
#pragma unroll
            for (int r = 0; r < 8; ++r) {
                int m = m0 + wm*64 + i*16 + r + 8*half;
                int n = n0 + wn*32 + j*16 + lrow;
                C[(size_t)m * N + n] = (TC)acc[i][j][r];
            }
}

// ---------------------------------------------------------------------------
// Interleaved RoPE, fp32 -> bf16, optional scale (q gets 1/sqrt(64)).
// Layout: [2048][nheads][64], one thread per (t, head, pair).
// ---------------------------------------------------------------------------
__global__ __launch_bounds__(256) void rope_kernel(
    const float* __restrict__ src, __bf16* __restrict__ dst, int nheads, float scale)
{
    int idx  = blockIdx.x * blockDim.x + threadIdx.x;
    int i    = idx & 31;
    int head = (idx >> 5) % nheads;
    int t    = idx / (nheads * 32);
    if (t >= 2048) return;
    float theta    = (float)(2 * i) * (1.0f / 64.0f);
    float inv_freq = 1.0f / (__powf(10000.0f, theta) + 1e-8f);
    float ang      = (float)t * inv_freq;
    float c = __cosf(ang), s = __sinf(ang);
    size_t base = (size_t)t * (nheads * 64) + head * 64 + 2 * i;
    float x1 = src[base], x2 = src[base + 1];
    dst[base]     = (__bf16)((x1 * c - x2 * s) * scale);
    dst[base + 1] = (__bf16)((x1 * s + x2 * c) * scale);
}

// ---------------------------------------------------------------------------
// lambda_full = exp(sum lq1*lk1) - exp(sum lq2*lk2) + LAMBDA_INIT
// ---------------------------------------------------------------------------
__global__ void lambda_kernel(const float* __restrict__ lq1, const float* __restrict__ lk1,
                              const float* __restrict__ lq2, const float* __restrict__ lk2,
                              float* __restrict__ out)
{
    if (threadIdx.x == 0) {
        float s1 = 0.f, s2 = 0.f;
        for (int i = 0; i < 64; ++i) { s1 += lq1[i] * lk1[i]; s2 += lq2[i] * lk2[i]; }
        *out = __expf(s1) - __expf(s2) + LAMBDA_INIT_F;
    }
}

// ---------------------------------------------------------------------------
// Fused flash differential attention.
// Grid: (32 q-tiles, 16 output heads). Block: 256 threads = 8 waves.
//   waves 0..3: q-head 2h (m-blocks 0..3), waves 4..7: q-head 2h+1.
// Both q-heads of a pair share K head h; V head = h/2 (128-dim).
// Online softmax; O = O0/l0 - lambda*O1/l1; RMS-norm; * subln * (1-lambda_init).
// K tiles staged with GLOBAL_LOAD_ASYNC_TO_LDS_B128 (ASYNCcnt path).
// ---------------------------------------------------------------------------
__global__ __launch_bounds__(256) void diff_attn(
    const __bf16* __restrict__ qp,   // [2048][32*64]  (rope'd, pre-scaled)
    const __bf16* __restrict__ kp,   // [2048][16*64]  (rope'd)
    const __bf16* __restrict__ vp,   // [2048][8*128]
    const float*  __restrict__ lam_p,
    const float*  __restrict__ subln,  // [128]
    __bf16* __restrict__ out)        // [2048][16*128]
{
    extern __shared__ __align__(32) char smem[];
    __bf16* Kt   = (__bf16*)smem;                 // [64 key][64 dim]     8 KB
    __bf16* Vt   = (__bf16*)(smem + 8192);        // [128 dim][64 key]   16 KB
    float*  Sb   = (float*) (smem + 24576);       // [2][64][64] scores  32 KB (aliased as O)
    __bf16* Pb   = (__bf16*)(smem + 57344);       // [2][64][64] probs   16 KB
    float*  mrun = (float*) (smem + 73728);       // [128]
    float*  lrun = mrun + 128;                    // [128]
    float*  scl  = mrun + 256;                    // [128]

    const int h    = blockIdx.y;
    const int q0   = blockIdx.x * 64;
    const int tid  = threadIdx.x;
    const int lane = tid & 31;
    const int w    = tid >> 5;
    const int lrow = lane & 15, half = lane >> 4;
    const int whead = w >> 2, mb = w & 3;
    const float lam = *lam_p;

    if (tid < 128) { mrun[tid] = -1e30f; lrun[tid] = 0.f; }

    // Q fragments live in registers for the whole kernel (A-matrix layout).
    v16bf qf[2];
    {
        const __bf16* qrow = qp + (size_t)(q0 + mb*16 + lrow) * 2048 + (2*h + whead) * 64;
#pragma unroll
        for (int kf = 0; kf < 2; ++kf) {
            const __bf16* p = qrow + kf*32 + half*16;
            qf[kf] = cat16(ld8(p), ld8(p + 8));
        }
    }

    v8f zero = {};
    v8f acc_o[8];                    // 64 q-rows(16 per wave) x 128 v-dims
#pragma unroll
    for (int i = 0; i < 8; ++i) acc_o[i] = zero;

    // Per-thread staging coordinates.
    const int kkey = tid >> 2, kd0 = (tid & 3) * 16;      // K tile: 32B per thread
    const int vkey = tid & 63, vd0 = (tid >> 6) * 32;     // V tile: 64B per thread

    for (int s0 = 0; s0 < 2048; s0 += 64) {
        // Stage K tile [key][dim] via async LDS DMA (no VGPR round trip).
        {
            const __bf16* src = kp + (size_t)(s0 + kkey) * 1024 + h * 64 + kd0;
            unsigned int lds = (unsigned int)(uintptr_t)(Kt + kkey*64 + kd0);
            async_b128(lds, src);
            async_b128_o16(lds, src);
        }
        // Stage V tile transposed -> Vt[dim][key] (needs 2B-granular scatter).
        {
            const __bf16* src = vp + (size_t)(s0 + vkey) * 1024 + (h >> 1) * 128 + vd0;
#pragma unroll
            for (int c = 0; c < 4; ++c) {
                v8bf x = ld8(src + c*8);
#pragma unroll
                for (int i = 0; i < 8; ++i) Vt[(vd0 + c*8 + i)*64 + vkey] = x[i];
            }
        }
        wait_async0();
        __syncthreads();

        // S = Q * K^T  (per wave: 16 rows x 64 keys).
        // Batch all 8 B-fragments, then 8 WMMAs (4 independent acc chains).
        {
            v16bf kfr[4][2];
#pragma unroll
            for (int nt = 0; nt < 4; ++nt)
#pragma unroll
                for (int kf = 0; kf < 2; ++kf)
                    kfr[nt][kf] = *(const v16bf*)(Kt + (nt*16 + lrow)*64 + kf*32 + half*16);

            v8f sacc[4];
#pragma unroll
            for (int nt = 0; nt < 4; ++nt)
                sacc[nt] = __builtin_amdgcn_wmma_f32_16x16x32_bf16(
                        false, qf[0], false, kfr[nt][0], (short)0, zero, false, false);
#pragma unroll
            for (int nt = 0; nt < 4; ++nt)
                sacc[nt] = __builtin_amdgcn_wmma_f32_16x16x32_bf16(
                        false, qf[1], false, kfr[nt][1], (short)0, sacc[nt], false, false);
#pragma unroll
            for (int nt = 0; nt < 4; ++nt)
#pragma unroll
                for (int r = 0; r < 8; ++r) {
                    int m = mb*16 + r + 8*half;
                    Sb[whead*4096 + m*64 + nt*16 + lrow] = sacc[nt][r];
                }
        }
        __syncthreads();

        // Online softmax: one thread per row (128 rows = 2 heads x 64 queries).
        if (tid < 128) {
            float* srow = Sb + tid * 64;
            float tm = srow[0];
#pragma unroll 8
            for (int j = 1; j < 64; ++j) tm = fmaxf(tm, srow[j]);
            float mo = mrun[tid];
            float mn = fmaxf(mo, tm);
            float sc = __expf(mo - mn);
            float ssum = 0.f;
            __bf16* prow = Pb + tid * 64;
#pragma unroll 8
            for (int j = 0; j < 64; ++j) {
                float p = __expf(srow[j] - mn);
                ssum += p;
                prow[j] = (__bf16)p;
            }
            lrun[tid] = lrun[tid] * sc + ssum;
            mrun[tid] = mn;
            scl[tid]  = sc;
        }
        __syncthreads();

        // Rescale running output accumulators by per-row correction.
#pragma unroll
        for (int r = 0; r < 8; ++r) {
            float sc = scl[whead*64 + mb*16 + r + 8*half];
#pragma unroll
            for (int nt = 0; nt < 8; ++nt) acc_o[nt][r] *= sc;
        }

        // O += P * V   (per wave: 16 rows x 128 dims).
        v16bf pf[2];
        {
            const __bf16* prow = Pb + (size_t)(whead*64 + mb*16 + lrow) * 64;
#pragma unroll
            for (int kf = 0; kf < 2; ++kf) {
                const __bf16* p = prow + kf*32 + half*16;
                pf[kf] = cat16(ld8(p), ld8(p + 8));
            }
        }
#pragma unroll
        for (int kf = 0; kf < 2; ++kf) {
            v16bf vfr[8];
#pragma unroll
            for (int nt = 0; nt < 8; ++nt)
                vfr[nt] = *(const v16bf*)(Vt + (nt*16 + lrow)*64 + kf*32 + half*16);
#pragma unroll
            for (int nt = 0; nt < 8; ++nt)
                acc_o[nt] = __builtin_amdgcn_wmma_f32_16x16x32_bf16(
                        false, pf[kf], false, vfr[nt], (short)0, acc_o[nt], false, false);
        }
        __syncthreads();   // protect Kt/Vt/Sb/Pb before next tile
    }

    // Finalize: rn = 1/l per row.
    if (tid < 128) scl[tid] = 1.f / lrun[tid];
    __syncthreads();

    float* Ob = Sb;   // alias dead score buffer: [64 q][128 dim] fp32
    if (whead == 0) {
#pragma unroll
        for (int nt = 0; nt < 8; ++nt)
#pragma unroll
            for (int r = 0; r < 8; ++r) {
                int m = mb*16 + r + 8*half;
                Ob[m*128 + nt*16 + lrow] = acc_o[nt][r] * scl[m];
            }
    }
    __syncthreads();
    if (whead == 1) {
#pragma unroll
        for (int nt = 0; nt < 8; ++nt)
#pragma unroll
            for (int r = 0; r < 8; ++r) {
                int m = mb*16 + r + 8*half;
                Ob[m*128 + nt*16 + lrow] -= lam * acc_o[nt][r] * scl[64 + m];
            }
    }
    __syncthreads();

    // RMS-norm over 128 dims + subln scale + (1-lambda_init); write bf16.
    if (tid < 64) {
        const float osc = 1.f - LAMBDA_INIT_F;
        float* orow = Ob + tid * 128;
        float ss = 0.f;
#pragma unroll 8
        for (int j = 0; j < 128; ++j) ss += orow[j] * orow[j];
        float rms = rsqrtf(ss * (1.f / 128.f) + 1e-5f);
        __bf16* dst = out + (size_t)(q0 + tid) * 2048 + h * 128;
#pragma unroll 8
        for (int j = 0; j < 128; ++j)
            dst[j] = (__bf16)(orow[j] * rms * subln[j] * osc);
    }
}

// ---------------------------------------------------------------------------
// Host launcher.
// ---------------------------------------------------------------------------
extern "C" void kernel_launch(void* const* d_in, const int* in_sizes, int n_in,
                              void* d_out, int out_size, void* d_ws, size_t ws_size,
                              hipStream_t stream)
{
    (void)in_sizes; (void)n_in; (void)out_size; (void)ws_size;

    const float* q     = (const float*)d_in[0];
    const float* k     = (const float*)d_in[1];
    const float* v     = (const float*)d_in[2];
    const float* Wq    = (const float*)d_in[3];
    const float* Wk    = (const float*)d_in[4];
    const float* Wv    = (const float*)d_in[5];
    const float* Wout  = (const float*)d_in[6];
    const float* lq1   = (const float*)d_in[7];
    const float* lk1   = (const float*)d_in[8];
    const float* lq2   = (const float*)d_in[9];
    const float* lk2   = (const float*)d_in[10];
    const float* subln = (const float*)d_in[11];

    char* ws = (char*)d_ws;
    float*  qp_f = (float*) (ws + (size_t)0);            // 16 MB  (2048x2048 f32)
    float*  kp_f = (float*) (ws + (size_t)(16u << 20));  //  8 MB  (2048x1024 f32)
    __bf16* qp_b = (__bf16*)(ws + (size_t)(24u << 20));  //  8 MB
    __bf16* kp_b = (__bf16*)(ws + (size_t)(32u << 20));  //  4 MB
    __bf16* vp_b = (__bf16*)(ws + (size_t)(36u << 20));  //  4 MB
    __bf16* at_b = (__bf16*)(ws + (size_t)(40u << 20));  //  8 MB
    float*  lamb = (float*) (ws + (size_t)(48u << 20));  //  4 B

    dim3 blk(256);

    // Projections (bf16 WMMA, fp32 out for rope; v straight to bf16).
    gemm_bf16_wmma<float, float> <<<dim3(16, 16), blk, 0, stream>>>(q, Wq, qp_f, 2048, 2048, 2048);
    gemm_bf16_wmma<float, float> <<<dim3( 8, 16), blk, 0, stream>>>(k, Wk, kp_f, 2048, 1024, 2048);
    gemm_bf16_wmma<float, __bf16><<<dim3( 8, 16), blk, 0, stream>>>(v, Wv, vp_b, 2048, 1024, 2048);

    // RoPE (+ q scaling) -> bf16 operands for attention.
    rope_kernel<<<8192, blk, 0, stream>>>(qp_f, qp_b, 32, SCALING_F);
    rope_kernel<<<4096, blk, 0, stream>>>(kp_f, kp_b, 16, 1.0f);

    lambda_kernel<<<1, 32, 0, stream>>>(lq1, lk1, lq2, lk2, lamb);

    // Fused flash differential attention (74 KB dynamic LDS, 320 KB WGP budget).
    diff_attn<<<dim3(32, 16), blk, 75264, stream>>>(qp_b, kp_b, vp_b, lamb, subln, at_b);

    // Output projection -> fp32 d_out.
    gemm_bf16_wmma<__bf16, float><<<dim3(16, 16), blk, 0, stream>>>(at_b, Wout, (float*)d_out, 2048, 2048, 2048);
}
